// MultiHeadAttention_72825465471494
// MI455X (gfx1250) — matrix operand order repaired and checked by reading, not compiled, not run on hardware
//
#include <hip/hip_runtime.h>

// ---------------------------------------------------------------------------
// MI455X (gfx1250, wave32) multi-head causal attention, bf16 WMMA pipeline.
//   stage 1: qkv = X @ Wqkv^T + b   (scatter to [B,H,S,D])
//   stage 2: causal flash attention per (b,h,qtile)
//   stage 3: out = ctx @ Wout^T + b
// All LDS tiles are laid out so every WMMA fragment is 2x ds_load_b128.
// ---------------------------------------------------------------------------

typedef __attribute__((ext_vector_type(16))) __bf16 v16bf;
typedef __attribute__((ext_vector_type(8)))  __bf16 v8bf;
typedef __attribute__((ext_vector_type(4)))  __bf16 v4bf;
typedef __attribute__((ext_vector_type(8)))  float  v8f;

#define WMMA_BF16(a, b, c) \
  __builtin_amdgcn_wmma_f32_16x16x32_bf16(false, (a), false, (b), (short)0, (c), false, false)

// Two contiguous 8-element (16B) LDS loads -> one 16-element fragment.
static __device__ __forceinline__ v16bf ld_frag(const __bf16* p, int e0, int e1) {
  const v8bf lo = *(const v8bf*)(p + e0);
  const v8bf hi = *(const v8bf*)(p + e1);
  return __builtin_shufflevector(lo, hi, 0, 1, 2, 3, 4, 5, 6, 7, 8, 9, 10, 11, 12, 13, 14, 15);
}

// float4 -> packed bf16x4 LDS store (ds_store_b64).
static __device__ __forceinline__ void st4(__bf16* dst, float4 f) {
  v4bf t;
  t[0] = (__bf16)f.x; t[1] = (__bf16)f.y; t[2] = (__bf16)f.z; t[3] = (__bf16)f.w;
  *(v4bf*)dst = t;
}

// Reductions across the 16-lane half sharing one WMMA C row.
static __device__ __forceinline__ float half16_max(float x) {
#pragma unroll
  for (int off = 1; off < 16; off <<= 1) x = fmaxf(x, __shfl_xor(x, off, 32));
  return x;
}
static __device__ __forceinline__ float half16_sum(float x) {
#pragma unroll
  for (int off = 1; off < 16; off <<= 1) x += __shfl_xor(x, off, 32);
  return x;
}

// ---------------------------------------------------------------------------
// GEMM: C[M,N] = A[M,K] @ W[N,K]^T + bias[N].  Block 128 thr (4 waves).
// Block tile 64x128, BK=32; each wave computes 32x64 = 2x4 WMMA tiles.
// MODE 0: plain store; MODE 1: scatter qkv columns to [B,H,S,D] Q/K/V.
// LDS rows: 40 bf16 (5 v8bf chunks, 80B) -> 16B-aligned, bank-staggered.
// ---------------------------------------------------------------------------
template <int MODE>
__global__ __launch_bounds__(128) void gemm_bf16_kernel(
    const float* __restrict__ A, const float* __restrict__ W,
    const float* __restrict__ bias, float* __restrict__ C,
    float* __restrict__ Qo, float* __restrict__ Ko, float* __restrict__ Vo,
    int M, int N, int K) {
  __shared__ v8bf As[64][5];    // 64 rows x 32 K (stride 40)
  __shared__ v8bf Ws[128][5];   // 128 cols x 32 K (stride 40)
  __bf16* as = (__bf16*)As;
  __bf16* ws = (__bf16*)Ws;

  const int tid  = threadIdx.x;
  const int lane = tid & 31;
  const int wv   = tid >> 5;
  const int m0   = blockIdx.y * 64;
  const int n0   = blockIdx.x * 128;
  const int wm   = (wv >> 1) * 32;  // 0 / 32
  const int wn   = (wv & 1) * 64;   // 0 / 64

  const int mrow = lane & 15;
  const int half = lane >> 4;
  const int khiA = half * 8;
  const int kbB  = half * 16;

  // fragment base pointers (loop-invariant)
  const __bf16* afp[2] = {as + (wm + mrow) * 40, as + (wm + 16 + mrow) * 40};
  const __bf16* bfp[4] = {ws + (wn + mrow) * 40,      ws + (wn + 16 + mrow) * 40,
                          ws + (wn + 32 + mrow) * 40, ws + (wn + 48 + mrow) * 40};

  v8f acc[2][4];
#pragma unroll
  for (int i = 0; i < 2; ++i)
#pragma unroll
    for (int j = 0; j < 4; ++j) acc[i][j] = {};

  const int arow = tid >> 1, acol = (tid & 1) * 16;  // A stage: 16 floats/thr
  const int wrow = tid;                              // W stage: 32 floats/thr

  for (int k0 = 0; k0 < K; k0 += 32) {
    const float* ap = A + (size_t)(m0 + arow) * K + k0 + acol;
#pragma unroll
    for (int i = 0; i < 16; i += 4)
      st4(as + arow * 40 + acol + i, *(const float4*)(ap + i));
    const float* wp = W + (size_t)(n0 + wrow) * K + k0;
#pragma unroll
    for (int i = 0; i < 32; i += 4)
      st4(ws + wrow * 40 + i, *(const float4*)(wp + i));
    __syncthreads();

    v16bf af[2], bf[4];
#pragma unroll
    for (int s = 0; s < 2; ++s) af[s] = ld_frag(afp[s], khiA, 16 + khiA);
#pragma unroll
    for (int j = 0; j < 4; ++j) bf[j] = ld_frag(bfp[j], khiA, 16 + khiA);
#pragma unroll
    for (int i = 0; i < 2; ++i)
#pragma unroll
      for (int j = 0; j < 4; ++j) acc[i][j] = WMMA_BF16(af[i], bf[j], acc[i][j]);
    __syncthreads();
  }

#pragma unroll
  for (int i = 0; i < 2; ++i) {
#pragma unroll
    for (int j = 0; j < 4; ++j) {
#pragma unroll
      for (int r = 0; r < 8; ++r) {
        const int gm = m0 + wm + i * 16 + half * 8 + r;
        const int gn = n0 + wn + j * 16 + (lane & 15);
        const float v = acc[i][j][r] + bias[gn];
        if (MODE == 0) {
          C[(size_t)gm * N + gn] = v;
        } else {
          const int p = gn >> 10, e = gn & 1023;
          const int h = e >> 6, d = e & 63;
          const int b = gm >> 11, s = gm & 2047;
          float* dst = (p == 0) ? Qo : (p == 1) ? Ko : Vo;
          dst[(((size_t)(b * 16 + h) * 2048 + s) * 64 + d)] = v;
        }
      }
    }
  }
}

// ---------------------------------------------------------------------------
// Flash attention: grid = (S/64, B*H), block 128 thr (4 waves).
// Wave w owns query rows [qt*64+16w, +16).  Per 32-key tile:
//   4 WMMA (QK^T) + online softmax + 4 WMMA (P@V).
// Ks: [key][d] (stride 72), Vt: [d][key] (stride 40, transposed at staging),
// Ps: per-wave 16x32 (stride 40) -- all fragments 2x ds_load_b128.
// Mask generation is skipped (wave-uniform branch) for tiles fully below
// the causal diagonal.
// ---------------------------------------------------------------------------
__global__ __launch_bounds__(128) void flash_attn_kernel(
    const float* __restrict__ Q, const float* __restrict__ Kt,
    const float* __restrict__ V, float* __restrict__ ctx) {
  __shared__ v8bf KsV[32][9];      // 32 keys x 64 d (stride 72)
  __shared__ v8bf VtV[64][5];      // 64 d x 32 keys (stride 40)
  __shared__ v8bf PsV[4][16][5];   // per-wave 16 q x 32 keys (stride 40)
  __bf16* Ks = (__bf16*)KsV;
  __bf16* Vt = (__bf16*)VtV;
  __bf16* Ps = (__bf16*)PsV;

  const int tid  = threadIdx.x;
  const int lane = tid & 31;
  const int wv   = tid >> 5;
  const int qt   = blockIdx.x;
  const int bh   = blockIdx.y;
  const int b    = bh >> 4, h = bh & 15;

  const size_t head_off = (size_t)bh * 2048 * 64;
  const float* Qb = Q + head_off;
  const float* Kb = Kt + head_off;
  const float* Vb = V + head_off;

  const int qrow0 = qt * 64 + wv * 16;
  const int mrow  = lane & 15;
  const int half  = lane >> 4;
  const int khi   = half * 8;
  __bf16* Pw = Ps + wv * 16 * 40;

  // loop-invariant fragment bases
  const __bf16* kfr0 = Ks + mrow * 72;         // sub 0 keys
  const __bf16* kfr1 = Ks + (16 + mrow) * 72;  // sub 1 keys
  const __bf16* paB  = Pw + mrow * 40;
  const __bf16* vfp[4] = {Vt + mrow * 40,        Vt + (16 + mrow) * 40,
                          Vt + (32 + mrow) * 40, Vt + (48 + mrow) * 40};

  // Q fragments (A layout), vector global loads, pre-scaled by 1/sqrt(64).
  v16bf qf[2];
  {
    const float* qp = Qb + (size_t)(qrow0 + mrow) * 64;
#pragma unroll
    for (int ks = 0; ks < 2; ++ks) {
      const float* bp = qp + ks * 32 + khi;
      float4 f0 = *(const float4*)(bp);
      float4 f1 = *(const float4*)(bp + 4);
      float4 f2 = *(const float4*)(bp + 16);
      float4 f3 = *(const float4*)(bp + 20);
      qf[ks][0]  = (__bf16)(f0.x * 0.125f); qf[ks][1]  = (__bf16)(f0.y * 0.125f);
      qf[ks][2]  = (__bf16)(f0.z * 0.125f); qf[ks][3]  = (__bf16)(f0.w * 0.125f);
      qf[ks][4]  = (__bf16)(f1.x * 0.125f); qf[ks][5]  = (__bf16)(f1.y * 0.125f);
      qf[ks][6]  = (__bf16)(f1.z * 0.125f); qf[ks][7]  = (__bf16)(f1.w * 0.125f);
      qf[ks][8]  = (__bf16)(f2.x * 0.125f); qf[ks][9]  = (__bf16)(f2.y * 0.125f);
      qf[ks][10] = (__bf16)(f2.z * 0.125f); qf[ks][11] = (__bf16)(f2.w * 0.125f);
      qf[ks][12] = (__bf16)(f3.x * 0.125f); qf[ks][13] = (__bf16)(f3.y * 0.125f);
      qf[ks][14] = (__bf16)(f3.z * 0.125f); qf[ks][15] = (__bf16)(f3.w * 0.125f);
    }
  }

  v8f o[4];
#pragma unroll
  for (int dt = 0; dt < 4; ++dt) o[dt] = {};
  float mst[8], lst[8];
#pragma unroll
  for (int r = 0; r < 8; ++r) { mst[r] = -1e30f; lst[r] = 0.0f; }

  const int lrow = tid >> 2;        // key 0..31
  const int lcol = (tid & 3) * 16;  // d base 0..48
  const int nkt  = 2 * qt + 2;      // causal key-tile count

  for (int kt = 0; kt < nkt; ++kt) {
    const int k0 = kt * 32;
    const float* kp = Kb + (size_t)(k0 + lrow) * 64 + lcol;
    const float* vp = Vb + (size_t)(k0 + lrow) * 64 + lcol;
    if (kt + 1 < nkt) {  // next tile toward L2 (global_prefetch_b8)
      __builtin_prefetch(kp + 32 * 64, 0, 1);
      __builtin_prefetch(vp + 32 * 64, 0, 1);
    }
#pragma unroll
    for (int i = 0; i < 16; i += 4) {
      st4(Ks + lrow * 72 + lcol + i, *(const float4*)(kp + i));   // K: [key][d]
      float4 fv = *(const float4*)(vp + i);                        // V: [d][key]
      Vt[(lcol + i + 0) * 40 + lrow] = (__bf16)fv.x;
      Vt[(lcol + i + 1) * 40 + lrow] = (__bf16)fv.y;
      Vt[(lcol + i + 2) * 40 + lrow] = (__bf16)fv.z;
      Vt[(lcol + i + 3) * 40 + lrow] = (__bf16)fv.w;
    }
    __syncthreads();

    // All four K fragments first, then the WMMA chain (partial dscnt waits).
    v16bf kf00 = ld_frag(kfr0, khi, 16 + khi);
    v16bf kf01 = ld_frag(kfr0, 32 + khi, 48 + khi);
    v16bf kf10 = ld_frag(kfr1, khi, 16 + khi);
    v16bf kf11 = ld_frag(kfr1, 32 + khi, 48 + khi);
    v8f sc0 = {}, sc1 = {};
    sc0 = WMMA_BF16(qf[0], kf00, sc0);
    sc1 = WMMA_BF16(qf[0], kf10, sc1);
    sc0 = WMMA_BF16(qf[1], kf01, sc0);
    sc1 = WMMA_BF16(qf[1], kf11, sc1);

    // Wave-uniform: tiles fully below the diagonal need no mask.
    if (k0 + 31 > qrow0) {
#pragma unroll
      for (int r = 0; r < 8; ++r) {
        const int qg = qrow0 + half * 8 + r;
        const int kg = k0 + (lane & 15);
        if (kg > qg)      sc0[r] = -1e30f;
        if (kg + 16 > qg) sc1[r] = -1e30f;
      }
    }

    // online softmax (C layout: row = r + 8*half, col = lane&15)
    float alpha[8];
#pragma unroll
    for (int r = 0; r < 8; ++r) {
      const float mx = half16_max(fmaxf(sc0[r], sc1[r]));
      const float mn = fmaxf(mst[r], mx);
      const float a  = __expf(mst[r] - mn);
      const float p0 = __expf(sc0[r] - mn);
      const float p1 = __expf(sc1[r] - mn);
      Pw[(half * 8 + r) * 40 + (lane & 15)]      = (__bf16)p0;
      Pw[(half * 8 + r) * 40 + 16 + (lane & 15)] = (__bf16)p1;
      lst[r] = lst[r] * a + half16_sum(p0 + p1);
      mst[r] = mn;
      alpha[r] = a;
    }

#pragma unroll
    for (int dt = 0; dt < 4; ++dt)
#pragma unroll
      for (int r = 0; r < 8; ++r) o[dt][r] *= alpha[r];

    // P@V: load all fragments, then the WMMA chain
    v16bf pa = ld_frag(paB, khi, 16 + khi);
    v16bf vf[4];
#pragma unroll
    for (int dt = 0; dt < 4; ++dt) vf[dt] = ld_frag(vfp[dt], khi, 16 + khi);
#pragma unroll
    for (int dt = 0; dt < 4; ++dt) o[dt] = WMMA_BF16(pa, vf[dt], o[dt]);
    __syncthreads();
  }

  // normalize + store ctx as [B,S,E], E index = h*64 + d
  float inv[8];
#pragma unroll
  for (int r = 0; r < 8; ++r) inv[r] = 1.0f / lst[r];
#pragma unroll
  for (int dt = 0; dt < 4; ++dt) {
#pragma unroll
    for (int r = 0; r < 8; ++r) {
      const int qg = qrow0 + half * 8 + r;
      const int e  = h * 64 + dt * 16 + (lane & 15);
      ctx[((size_t)(b * 2048 + qg)) * 1024 + e] = o[dt][r] * inv[r];
    }
  }
}

// ---------------------------------------------------------------------------
extern "C" void kernel_launch(void* const* d_in, const int* in_sizes, int n_in,
                              void* d_out, int out_size, void* d_ws, size_t ws_size,
                              hipStream_t stream) {
  const float* X    = (const float*)d_in[0];  // [2,2048,1024]
  const float* Wqkv = (const float*)d_in[1];  // [3072,1024]
  const float* Bqkv = (const float*)d_in[2];  // [3072]
  const float* Wout = (const float*)d_in[3];  // [1024,1024]
  const float* Bout = (const float*)d_in[4];  // [1024]
  float* out = (float*)d_out;                 // [2,2048,1024]

  float* ws = (float*)d_ws;
  const size_t PER = (size_t)2 * 16 * 2048 * 64;  // 4M floats / tensor
  float* Qb  = ws;
  float* Kb  = ws + PER;
  float* Vb  = ws + 2 * PER;
  float* Ctx = ws + 3 * PER;

  const int M = 4096, K = 1024;
  dim3 blk(128);

  gemm_bf16_kernel<1><<<dim3(3072 / 128, M / 64), blk, 0, stream>>>(
      X, Wqkv, Bqkv, nullptr, Qb, Kb, Vb, M, 3072, K);

  flash_attn_kernel<<<dim3(2048 / 64, 2 * 16), blk, 0, stream>>>(Qb, Kb, Vb, Ctx);

  gemm_bf16_kernel<0><<<dim3(1024 / 128, M / 64), blk, 0, stream>>>(
      Ctx, Wout, Bout, out, nullptr, nullptr, nullptr, M, 1024, K);
}
